// TransposedAttention_32452772889002
// MI455X (gfx1250) — compile-verified, matrix operand order checked by reading, not run on hardware
//
#include <hip/hip_runtime.h>

// ---------------------------------------------------------------------------
// TransposedAttention for MI455X (gfx1250, wave32, WMMA + TDM).
//
// Algebraic rewrite:  dots = Wq^T (x^T x) Wk,  vv = Wv^T (x^T x) Wv2.
// Pipeline:
//   1) convert W_qkv, W_out fp32 -> f16 (workspace)
//   2) gram:    G[b]   = x[b]^T x[b]                  (f16 out, fp32 acc WMMA)
//   3) project: P[b]   = G[b] @ [Wk_all | Wv2_all]    (TDM A-tiles, dbl-buffer)
//   4) inner:   per (b,h): S = Wq^T P_k, T = Wv^T P_v2, softmax(S/8) @ T
//   5) final:   out = out_mid @ W_out + b_out         (TDM A-tiles, dbl-buffer)
//
// LDS tiles are fragment-major [row][k]: one WMMA fragment = 2x ds_load_b128.
// Staging transposes pack two k-adjacent halves into 32-bit LDS stores.
// ---------------------------------------------------------------------------

typedef __attribute__((ext_vector_type(16))) _Float16 v16h;
typedef __attribute__((ext_vector_type(8)))  _Float16 v8h;
typedef __attribute__((ext_vector_type(8)))  float    v8f;

#define B_SZ   8
#define N_SEQ  4096
#define DIM    1024
#define HEADS  16

// ---------------------------------------------------------------------------
// Tensor Data Mover: 2-D f16 tile (tile_w elems x tile_h rows) global -> LDS,
// LDS rows padded to tile_w+8 halves via D# pad (interval 16 DW, amount 4 DW).
#if defined(__gfx1250__) && __has_builtin(__builtin_amdgcn_tensor_load_to_lds)
#define HAVE_TDM 1
typedef __attribute__((ext_vector_type(4))) unsigned int u32x4;
typedef __attribute__((ext_vector_type(8))) int          i32x8;
typedef __attribute__((ext_vector_type(4))) int          i32x4;

__device__ __forceinline__ void tdm_load_tile_2d(const _Float16* g,
                                                 unsigned lds_byte_addr,
                                                 unsigned tile_w, unsigned tile_h,
                                                 unsigned row_stride_elems) {
  unsigned long long ga = (unsigned long long)(size_t)g;
  u32x4 g0;
  g0[0] = 1u;                                   // count=1, user descriptor
  g0[1] = lds_byte_addr;                        // lds_addr
  g0[2] = (unsigned)(ga & 0xffffffffu);         // global_addr[31:0]
  g0[3] = (unsigned)((ga >> 32) & 0x01ffffffu)  // global_addr[56:32]
          | (2u << 30);                         // type = 2 ("image")
  const unsigned td0 = row_stride_elems;        // tensor dim0 (elems)
  const unsigned td1 = 1u << 20;                // tensor dim1: no clipping
  i32x8 g1;
  g1[0] = (int)((1u << 16)                      // data_size = 1 -> 2 bytes
                | (1u << 20)                    // pad_enable
                | (3u << 22)                    // pad_interval: 16 DWORDs
                | (3u << 25));                  // pad_amount:   4 DWORDs
  g1[1] = (int)((td0 & 0xffffu) << 16);                                // dim0 lo
  g1[2] = (int)(((td0 >> 16) & 0xffffu) | ((td1 & 0xffffu) << 16));    // dim0 hi | dim1 lo
  g1[3] = (int)(((td1 >> 16) & 0xffffu) | ((tile_w & 0xffffu) << 16)); // dim1 hi | tile_dim0
  g1[4] = (int)(tile_h & 0xffffu);                                     // tile_dim1
  g1[5] = (int)row_stride_elems;                                       // dim0_stride[31:0]
  g1[6] = 0;
  g1[7] = 0;
  i32x4 z4 = {0, 0, 0, 0};
#if __clang_major__ >= 23
  i32x8 z8 = {0, 0, 0, 0, 0, 0, 0, 0};
  __builtin_amdgcn_tensor_load_to_lds(g0, g1, z4, z4, z8, 0);
#else
  __builtin_amdgcn_tensor_load_to_lds(g0, g1, z4, z4, 0);
#endif
}
#endif

// K index runs for the 16-bit 16x32 WMMA fragment (CDNA5 ISA 7.12.2):
// lane<16: K = 0..7 and 16..23 ; lane>=16: K = 8..15 and 24..31.
// With tiles stored [row][k] (k contiguous), a fragment is two b128 loads.
__device__ __forceinline__ v16h frag_rowk(const _Float16* lds, int stride,
                                          int row0, int k0, int lane) {
  const _Float16* p = lds + (row0 + (lane & 15)) * stride + k0 + ((lane >> 4) << 3);
  union { v16h v; v8h h[2]; } u;
  u.h[0] = *(const v8h*)(p);
  u.h[1] = *(const v8h*)(p + 16);
  return u.v;
}

__device__ __forceinline__ v8f wmma16(v16h a, v16h b, v8f c) {
  return __builtin_amdgcn_wmma_f32_16x16x32_f16(false, a, false, b,
                                                (short)0, c, false, false);
}

// Pack two fp32 -> two f16 in one 32-bit LDS store (k-adjacent pair).
__device__ __forceinline__ void pack2f(_Float16* dst, float lo, float hi) {
  union { unsigned u; _Float16 h[2]; } p;
  p.h[0] = (_Float16)lo;
  p.h[1] = (_Float16)hi;
  *(unsigned*)dst = p.u;
}

// Transpose-scatter two global rows (r, r+1) of 8 halves at octet oc into a
// [col][k] LDS tile (stride 40): 2x global b128 -> 8x 32-bit LDS stores.
__device__ __forceinline__ void scatter_pair_rows(const _Float16* __restrict__ gsrc,
                                                  size_t ld, int r, int oc,
                                                  _Float16* dst) {
  union { uint4 u; _Float16 h[8]; } v0, v1;
  v0.u = *(const uint4*)(gsrc + (size_t)r * ld + oc * 8);
  v1.u = *(const uint4*)(gsrc + (size_t)(r + 1) * ld + oc * 8);
#pragma unroll
  for (int u = 0; u < 8; ++u) {
    union { unsigned uu; _Float16 h[2]; } p;
    p.h[0] = v0.h[u];
    p.h[1] = v1.h[u];
    *(unsigned*)(dst + (oc * 8 + u) * 40 + r) = p.uu;
  }
}

// ---------------------------------------------------------------------------
__global__ void convert_f32_f16(const float* __restrict__ in,
                                _Float16* __restrict__ out, int n) {
  int i = blockIdx.x * blockDim.x + threadIdx.x;
  int stride = gridDim.x * blockDim.x;
  for (; i < n; i += stride) out[i] = (_Float16)in[i];
}

// ---------------------------------------------------------------------------
// G[b] = x[b]^T x[b]  ([1024,1024]/batch).  Grid (8,8,B), 256 threads.
// Double-buffered LDS; tiles stored transposed ([col][k]) while cvt fp32->f16.
__global__ void gram_kernel(const float* __restrict__ x, _Float16* __restrict__ G) {
  __shared__ __attribute__((aligned(16))) _Float16 sA[2][128 * 40];
  __shared__ __attribute__((aligned(16))) _Float16 sB[2][128 * 40];
  const int b  = blockIdx.z;
  const int i0 = blockIdx.x * 128;
  const int j0 = blockIdx.y * 128;
  const int tid = threadIdx.x, lane = tid & 31, w = tid >> 5;
  const float* xb = x + (size_t)b * N_SEQ * DIM;

  auto stage = [&](int k0, int bufi) {
#pragma unroll
    for (int i = 0; i < 2; ++i) {
      int idx = tid + i * 256;          // 512 tasks: 16 row-pairs x 32 col-quads
      int rp = idx >> 5, c4 = idx & 31;
      int r = rp * 2, c = c4 * 4;
      const float* s0 = xb + (size_t)(k0 + r) * DIM;
      const float* s1 = s0 + DIM;
      float4 a0 = *(const float4*)(s0 + i0 + c);
      float4 a1 = *(const float4*)(s1 + i0 + c);
      float4 b0 = *(const float4*)(s0 + j0 + c);
      float4 b1 = *(const float4*)(s1 + j0 + c);
      _Float16* da = &sA[bufi][0];
      _Float16* db = &sB[bufi][0];
      pack2f(da + (c + 0) * 40 + r, a0.x, a1.x);
      pack2f(da + (c + 1) * 40 + r, a0.y, a1.y);
      pack2f(da + (c + 2) * 40 + r, a0.z, a1.z);
      pack2f(da + (c + 3) * 40 + r, a0.w, a1.w);
      pack2f(db + (c + 0) * 40 + r, b0.x, b1.x);
      pack2f(db + (c + 1) * 40 + r, b0.y, b1.y);
      pack2f(db + (c + 2) * 40 + r, b0.z, b1.z);
      pack2f(db + (c + 3) * 40 + r, b0.w, b1.w);
    }
  };

  v8f acc[8] = {};
  stage(0, 0);
  int buf = 0;
  for (int k0 = 0; k0 < N_SEQ; k0 += 32) {
    __syncthreads();                       // staging of current buffer done
    if (k0 + 32 < N_SEQ) stage(k0 + 32, buf ^ 1);
    if (k0 + 64 < N_SEQ)                   // global_prefetch_b8 L2 warm-up
      __builtin_prefetch(xb + (size_t)(k0 + 64 + (tid >> 3)) * DIM + i0 + (tid & 7) * 16, 0, 1);
    v16h a = frag_rowk(&sA[buf][0], 40, w * 16, 0, lane);
    v16h bf[8];
#pragma unroll
    for (int s = 0; s < 8; ++s) bf[s] = frag_rowk(&sB[buf][0], 40, s * 16, 0, lane);
#pragma unroll
    for (int s = 0; s < 8; ++s) acc[s] = wmma16(a, bf[s], acc[s]);
    buf ^= 1;
  }
  const int half = lane >> 4, nlo = lane & 15;
  _Float16* Gb = G + (size_t)b * DIM * DIM;
#pragma unroll
  for (int s = 0; s < 8; ++s)
#pragma unroll
    for (int r = 0; r < 8; ++r)
      Gb[(size_t)(i0 + w * 16 + r + 8 * half) * DIM + j0 + s * 16 + nlo] =
          (_Float16)acc[s][r];
}

// ---------------------------------------------------------------------------
// P[b] = G[b] @ [Wk_all | Wv2_all]  ([1024,2048]/batch). Grid (8,16,B).
// Double-buffered; A tile arrives via TDM overlapped with WMMA on cur buffer.
__global__ void project_kernel(const _Float16* __restrict__ G,
                               const _Float16* __restrict__ Wqkv,
                               _Float16* __restrict__ P) {
  __shared__ __attribute__((aligned(16))) _Float16 sA[2][128 * 40];  // G  [m][k]
  __shared__ __attribute__((aligned(16))) _Float16 sB[2][128 * 40];  // W2 [n][k]
  const int b  = blockIdx.z;
  const int m0 = blockIdx.x * 128;
  const int n0 = blockIdx.y * 128;
  const int wcol0 = (n0 < 1024) ? (1024 + n0) : (2048 + n0);  // k / v2 block
  const int tid = threadIdx.x, lane = tid & 31, w = tid >> 5;
  const _Float16* Gb = G + (size_t)b * DIM * DIM;

  auto stageA = [&](int k0, int bufi) {
#if HAVE_TDM
    if (tid < 32)  // one wave issues the descriptor; EXEC is ignored
      tdm_load_tile_2d(Gb + (size_t)m0 * DIM + k0,
                       (unsigned)(size_t)(void*)&sA[bufi][0], 32, 128, DIM);
#else
#pragma unroll
    for (int i = 0; i < 2; ++i) {
      int idx = tid + i * 256;
      int r = idx >> 2, oc = idx & 3;
      *(uint4*)(&sA[bufi][0] + r * 40 + oc * 8) =
          *(const uint4*)(Gb + (size_t)(m0 + r) * DIM + k0 + oc * 8);
    }
#endif
  };
  auto stageB = [&](int k0, int bufi) {
    int rp = tid >> 4, oc = tid & 15;  // 16 row-pairs x 16 octets
    scatter_pair_rows(Wqkv + (size_t)k0 * 4096 + wcol0, 4096, rp * 2, oc,
                      &sB[bufi][0]);
  };

  v8f acc[8] = {};
  stageA(0, 0);
  stageB(0, 0);
  int buf = 0;
  for (int k0 = 0; k0 < DIM; k0 += 32) {
#if HAVE_TDM
    if (tid < 32) __builtin_amdgcn_s_wait_tensorcnt(0);  // cur A tile landed
#endif
    __syncthreads();
    if (k0 + 32 < DIM) {
      stageA(k0 + 32, buf ^ 1);  // async TDM into next buffer
      stageB(k0 + 32, buf ^ 1);
    }
    v16h a = frag_rowk(&sA[buf][0], 40, w * 16, 0, lane);
    v16h bf[8];
#pragma unroll
    for (int s = 0; s < 8; ++s) bf[s] = frag_rowk(&sB[buf][0], 40, s * 16, 0, lane);
#pragma unroll
    for (int s = 0; s < 8; ++s) acc[s] = wmma16(a, bf[s], acc[s]);
    buf ^= 1;
  }
  const int half = lane >> 4, nlo = lane & 15;
  _Float16* Pb = P + (size_t)b * DIM * 2048;
#pragma unroll
  for (int s = 0; s < 8; ++s)
#pragma unroll
    for (int r = 0; r < 8; ++r)
      Pb[(size_t)(m0 + w * 16 + r + 8 * half) * 2048 + n0 + s * 16 + nlo] =
          (_Float16)acc[s][r];
}

// ---------------------------------------------------------------------------
// C[d,e] = sum_p gA[p,d]*gB[p,e]; slabs staged transposed -> [col][k].
__device__ __forceinline__ void gemm_64x64_redK(
    const _Float16* __restrict__ gA, int ldA,
    const _Float16* __restrict__ gB, int ldB,
    _Float16* sA, _Float16* sB, v8f acc[2],
    int sm0, int sn0, int sm1, int sn1, int lane, int tid) {
  const int slab = tid >> 7;           // waves 0-3: slab A, waves 4-7: slab B
  const int tt = tid & 127;
  const int r = (tt >> 3) * 2, oc = tt & 7;
  for (int p0 = 0; p0 < DIM; p0 += 32) {
    if (slab == 0)
      scatter_pair_rows(gA + (size_t)p0 * ldA, ldA, r, oc, sA);
    else
      scatter_pair_rows(gB + (size_t)p0 * ldB, ldB, r, oc, sB);
    __syncthreads();
    v16h a0 = frag_rowk(sA, 40, sm0 * 16, 0, lane);
    v16h b0 = frag_rowk(sB, 40, sn0 * 16, 0, lane);
    v16h a1 = frag_rowk(sA, 40, sm1 * 16, 0, lane);
    v16h b1 = frag_rowk(sB, 40, sn1 * 16, 0, lane);
    acc[0] = wmma16(a0, b0, acc[0]);
    acc[1] = wmma16(a1, b1, acc[1]);
    __syncthreads();
  }
}

// Per (b,h): S = Wq^T P_k / 8 ; T = Wv^T P_v2 ; out = softmax(S) @ T.
// Grid (HEADS, B), 256 threads (8 waves x 2 sub-tiles of the 4x4 grid).
__global__ void inner_kernel(const _Float16* __restrict__ Wqkv,
                             const _Float16* __restrict__ P,
                             _Float16* __restrict__ Om) {
  __shared__ __attribute__((aligned(16))) _Float16 sA[64 * 40];
  __shared__ __attribute__((aligned(16))) _Float16 sB[64 * 40];
  __shared__ __attribute__((aligned(16))) float    sS[64 * 68];
  __shared__ __attribute__((aligned(16))) _Float16 sT[64 * 72];   // [f][e]
  __shared__ __attribute__((aligned(16))) _Float16 sAt[64 * 72];  // [d][e]
  const int h = blockIdx.x, b = blockIdx.y;
  const int tid = threadIdx.x, lane = tid & 31, w = tid >> 5;
  const int id0 = w * 2, id1 = w * 2 + 1;
  const int sm0 = id0 >> 2, sn0 = id0 & 3;
  const int sm1 = id1 >> 2, sn1 = id1 & 3;
  const _Float16* Pb = P + (size_t)b * DIM * 2048;

  v8f sacc[2] = {}, tacc[2] = {};
  gemm_64x64_redK(Wqkv + h * 64, 4096, Pb + h * 64, 2048,
                  sA, sB, sacc, sm0, sn0, sm1, sn1, lane, tid);
  gemm_64x64_redK(Wqkv + 2048 + h * 64, 4096, Pb + 1024 + h * 64, 2048,
                  sA, sB, tacc, sm0, sn0, sm1, sn1, lane, tid);

  const int half = lane >> 4, nlo = lane & 15;
#pragma unroll
  for (int s = 0; s < 2; ++s) {
    const int sm = s ? sm1 : sm0, sn = s ? sn1 : sn0;
#pragma unroll
    for (int r = 0; r < 8; ++r) {
      int rr = sm * 16 + r + 8 * half, cc = sn * 16 + nlo;
      sS[rr * 68 + cc] = (s ? sacc[1][r] : sacc[0][r]) * 0.125f;    // d^-0.5
      sT[cc * 72 + rr] = (_Float16)(s ? tacc[1][r] : tacc[0][r]);   // [f][e]
    }
  }
  __syncthreads();
  if (tid < 64) {  // row-wise softmax over e
    float mx = -3.0e38f;
    for (int e = 0; e < 64; ++e) mx = fmaxf(mx, sS[tid * 68 + e]);
    float sum = 0.f;
    for (int e = 0; e < 64; ++e) {
      float ex = __expf(sS[tid * 68 + e] - mx);
      sS[tid * 68 + e] = ex;
      sum += ex;
    }
    float inv = 1.0f / sum;
    for (int e = 0; e < 64; ++e)
      sAt[tid * 72 + e] = (_Float16)(sS[tid * 68 + e] * inv);
  }
  __syncthreads();

  v8f oacc[2] = {};
#pragma unroll
  for (int e0 = 0; e0 < 64; e0 += 32) {
    v16h a0 = frag_rowk(sAt, 72, sm0 * 16, e0, lane);
    v16h b0 = frag_rowk(sT, 72, sn0 * 16, e0, lane);
    v16h a1 = frag_rowk(sAt, 72, sm1 * 16, e0, lane);
    v16h b1 = frag_rowk(sT, 72, sn1 * 16, e0, lane);
    oacc[0] = wmma16(a0, b0, oacc[0]);
    oacc[1] = wmma16(a1, b1, oacc[1]);
  }
  _Float16* Ob = Om + (size_t)b * 64 * DIM + h * 64;
#pragma unroll
  for (int s = 0; s < 2; ++s) {
    const int sm = s ? sm1 : sm0, sn = s ? sn1 : sn0;
#pragma unroll
    for (int r = 0; r < 8; ++r)
      Ob[(size_t)(sm * 16 + r + 8 * half) * DIM + sn * 16 + nlo] =
          (_Float16)(s ? oacc[1][r] : oacc[0][r]);
  }
}

// ---------------------------------------------------------------------------
// out[512,1024] = Om[512,1024] @ W_out + b_out.  Grid (8,8), 256 threads.
__global__ void final_kernel(const _Float16* __restrict__ Om,
                             const _Float16* __restrict__ Wo,
                             const float* __restrict__ bias,
                             float* __restrict__ out) {
  __shared__ __attribute__((aligned(16))) _Float16 sA[2][64 * 40];   // Om [m][k]
  __shared__ __attribute__((aligned(16))) _Float16 sB[2][128 * 40];  // Wo [n][k]
  const int m0 = blockIdx.x * 64;
  const int n0 = blockIdx.y * 128;
  const int tid = threadIdx.x, lane = tid & 31, w = tid >> 5;

  auto stageA = [&](int k0, int bufi) {
#if HAVE_TDM
    if (tid < 32)
      tdm_load_tile_2d(Om + (size_t)m0 * DIM + k0,
                       (unsigned)(size_t)(void*)&sA[bufi][0], 32, 64, DIM);
#else
    {
      int r = tid >> 2, oc = tid & 3;  // 64 rows x 4 uint4
      *(uint4*)(&sA[bufi][0] + r * 40 + oc * 8) =
          *(const uint4*)(Om + (size_t)(m0 + r) * DIM + k0 + oc * 8);
    }
#endif
  };
  auto stageB = [&](int k0, int bufi) {
    int rp = tid >> 4, oc = tid & 15;
    scatter_pair_rows(Wo + (size_t)k0 * DIM + n0, DIM, rp * 2, oc, &sB[bufi][0]);
  };

  v8f acc[4] = {};
  stageA(0, 0);
  stageB(0, 0);
  int buf = 0;
  for (int k0 = 0; k0 < DIM; k0 += 32) {
#if HAVE_TDM
    if (tid < 32) __builtin_amdgcn_s_wait_tensorcnt(0);
#endif
    __syncthreads();
    if (k0 + 32 < DIM) {
      stageA(k0 + 32, buf ^ 1);
      stageB(k0 + 32, buf ^ 1);
    }
    v16h bf = frag_rowk(&sB[buf][0], 40, w * 16, 0, lane);
    v16h af[4];
#pragma unroll
    for (int s = 0; s < 4; ++s) af[s] = frag_rowk(&sA[buf][0], 40, s * 16, 0, lane);
#pragma unroll
    for (int s = 0; s < 4; ++s) acc[s] = wmma16(af[s], bf, acc[s]);
    buf ^= 1;
  }
  const int half = lane >> 4, nlo = lane & 15;
  const int gc = n0 + w * 16 + nlo;
  const float bv = bias[gc];
#pragma unroll
  for (int s = 0; s < 4; ++s)
#pragma unroll
    for (int r = 0; r < 8; ++r)
      out[(size_t)(m0 + s * 16 + r + 8 * half) * DIM + gc] = acc[s][r] + bv;
}

// ---------------------------------------------------------------------------
extern "C" void kernel_launch(void* const* d_in, const int* in_sizes, int n_in,
                              void* d_out, int out_size, void* d_ws, size_t ws_size,
                              hipStream_t stream) {
  const float* x    = (const float*)d_in[0];   // [8,4096,1024]
  const float* Wqkv = (const float*)d_in[1];   // [1024,4096]
  const float* Wout = (const float*)d_in[2];   // [1024,1024]
  const float* bout = (const float*)d_in[3];   // [1024]

  char* ws = (char*)d_ws;
  const size_t MB = 1u << 20;
  _Float16* wqkv_h = (_Float16*)(ws + 0);        //  8 MB
  _Float16* wout_h = (_Float16*)(ws + 8 * MB);   //  2 MB
  _Float16* G      = (_Float16*)(ws + 10 * MB);  // 16 MB
  _Float16* P      = (_Float16*)(ws + 26 * MB);  // 32 MB
  _Float16* Om     = (_Float16*)(ws + 58 * MB);  //  1 MB (total 59 MB)

  convert_f32_f16<<<2048, 256, 0, stream>>>(Wqkv, wqkv_h, DIM * 4096);
  convert_f32_f16<<<512, 256, 0, stream>>>(Wout, wout_h, DIM * DIM);
  gram_kernel<<<dim3(8, 8, B_SZ), 256, 0, stream>>>(x, G);
  project_kernel<<<dim3(8, 16, B_SZ), 256, 0, stream>>>(G, wqkv_h, P);
  inner_kernel<<<dim3(HEADS, B_SZ), 256, 0, stream>>>(wqkv_h, P, Om);
  final_kernel<<<dim3(8, 8), 256, 0, stream>>>(Om, wout_h, bout, (float*)d_out);
}
